// PGATActor_84284438217479
// MI455X (gfx1250) — matrix operand order repaired and checked by reading, not compiled, not run on hardware
//
#include <hip/hip_runtime.h>
#include <hip/hip_bf16.h>
#include <stdint.h>

#define BB 512
#define NNODE 256
#define DIN 32
#define HID 128
#define HEADS 4
#define CC 32
#define KNB 5
#define NTOT (BB * NNODE)          /* 131072 nodes */
#define EDG (NTOT * (KNB + 1))     /* 786432 edges incl self-loops */

typedef __attribute__((ext_vector_type(16))) _Float16 v16h;
typedef __attribute__((ext_vector_type(8)))  float    v8f;

union HPK { _Float16 h[2]; uint32_t u; };
union AV  { v16h v; uint32_t u[8]; };

// ---------------- kNN graph build: one block per batch, one thread per node ----
__global__ void knn_kernel(const float* __restrict__ obs, int* __restrict__ nbr) {
    __shared__ float px[NNODE], py[NNODE];
    const int b = blockIdx.x, i = threadIdx.x;
    const float* base = obs + (size_t)b * NNODE * DIN;
    px[i] = base[i * DIN + 0];
    py[i] = base[i * DIN + 1];
    __syncthreads();
    const float xi = px[i], yi = py[i];
    float bd[KNB]; int bj[KNB];
#pragma unroll
    for (int k = 0; k < KNB; ++k) { bd[k] = 3.4e38f; bj[k] = 0; }
    for (int j = 0; j < NNODE; ++j) {
        if (j == i) continue;
        const float dx = xi - px[j], dy = yi - py[j];
        const float d = dx * dx + dy * dy;   // sqrt is monotone; ordering identical
        if (d < bd[KNB - 1]) {
            bd[KNB - 1] = d; bj[KNB - 1] = j;
#pragma unroll
            for (int s = KNB - 1; s > 0; --s) {
                if (bd[s] < bd[s - 1]) {
                    float td = bd[s]; bd[s] = bd[s - 1]; bd[s - 1] = td;
                    int   tj = bj[s]; bj[s] = bj[s - 1]; bj[s - 1] = tj;
                }
            }
        }
    }
    const int g = b * NNODE + i;
#pragma unroll
    for (int k = 0; k < KNB; ++k) nbr[g * KNB + k] = b * NNODE + bj[k];
}

// ---------------- reverse-CSR build ------------------------------------------
__global__ void degcount_kernel(const int* __restrict__ nbr, int* __restrict__ indeg) {
    const int i = blockIdx.x * 256 + threadIdx.x;
#pragma unroll
    for (int k = 0; k < KNB; ++k) atomicAdd(&indeg[nbr[i * KNB + k]], 1);
    atomicAdd(&indeg[i], 1);   // self-loop
}

__global__ void scan_block_kernel(const int* __restrict__ indeg,
                                  int* __restrict__ start, int* __restrict__ bsum) {
    __shared__ int sd[256];
    const int tid = threadIdx.x, g = blockIdx.x * 256 + tid;
    const int v = indeg[g];
    sd[tid] = v; __syncthreads();
    for (int off = 1; off < 256; off <<= 1) {
        int t = (tid >= off) ? sd[tid - off] : 0;
        __syncthreads();
        sd[tid] += t;
        __syncthreads();
    }
    start[g] = sd[tid] - v;                  // exclusive scan within block
    if (tid == 255) bsum[blockIdx.x] = sd[255];
}

__global__ void scan_top_kernel(const int* __restrict__ bsum, int* __restrict__ boff) {
    __shared__ int sd[512];
    const int tid = threadIdx.x;
    const int v = bsum[tid];
    sd[tid] = v; __syncthreads();
    for (int off = 1; off < 512; off <<= 1) {
        int t = (tid >= off) ? sd[tid - off] : 0;
        __syncthreads();
        sd[tid] += t;
        __syncthreads();
    }
    boff[tid] = sd[tid] - v;
}

__global__ void scan_add_kernel(int* __restrict__ start, const int* __restrict__ boff,
                                int* __restrict__ cursor) {
    const int g = blockIdx.x * 256 + threadIdx.x;
    const int s = start[g] + boff[blockIdx.x];
    start[g] = s;
    cursor[g] = s;
}

__global__ void fill_kernel(const int* __restrict__ nbr, int* __restrict__ cursor,
                            int* __restrict__ inedges) {
    const int i = blockIdx.x * 256 + threadIdx.x;
#pragma unroll
    for (int k = 0; k < KNB; ++k) {
        const int d = nbr[i * KNB + k];
        const int p = atomicAdd(&cursor[d], 1);
        inedges[p] = i;
    }
    const int p = atomicAdd(&cursor[i], 1);
    inedges[p] = i;
}

// ---------------- f32 -> f16 activation conversion ----------------------------
__global__ void cvt_kernel(const float* __restrict__ x, _Float16* __restrict__ y, int n) {
    const int i = blockIdx.x * 256 + threadIdx.x;
    if (i < n) y[i] = (_Float16)x[i];
}

// ---------------- pack W (KDIM x NOUT, f32) into WMMA B layout ----------------
// Layout: Wp[ks][n][kp] = f16 pair {W[ks*32+2kp][n], W[ks*32+2kp+1][n]}, n padded to NP.
template<int KDIM, int NOUT>
__global__ void packw_kernel(const float* __restrict__ W, uint32_t* __restrict__ Wp) {
    constexpr int NP = ((NOUT + 15) / 16) * 16;
    constexpr int KS = KDIM / 32;
    constexpr int TOTAL = KS * NP * 16;
    for (int idx = blockIdx.x * 256 + threadIdx.x; idx < TOTAL; idx += gridDim.x * 256) {
        const int ks = idx / (NP * 16);
        const int rem = idx - ks * NP * 16;
        const int n = rem >> 4, kp = rem & 15;
        const int k = ks * 32 + 2 * kp;
        float w0 = 0.f, w1 = 0.f;
        if (n < NOUT) {
            w0 = W[(size_t)k * NOUT + n];
            w1 = W[(size_t)(k + 1) * NOUT + n];
        }
        HPK p; p.h[0] = (_Float16)w0; p.h[1] = (_Float16)w1;
        Wp[idx] = p.u;
    }
}

// ---------------- attention logit projections (al_s, al_d) --------------------
__global__ void al_kernel(const float* __restrict__ h, const float* __restrict__ a_src,
                          const float* __restrict__ a_dst,
                          float* __restrict__ als, float* __restrict__ ald) {
    const int id = blockIdx.x * 256 + threadIdx.x;   // NTOT*HEADS threads
    const int n = id >> 2, hd = id & 3;
    const float* hp = h + (size_t)n * HID + hd * CC;
    const float* as = a_src + hd * CC;
    const float* ad = a_dst + hd * CC;
    float s = 0.f, d = 0.f;
#pragma unroll
    for (int c = 0; c < CC; ++c) { const float v = hp[c]; s += v * as[c]; d += v * ad[c]; }
    als[id] = s;
    ald[id] = d;
}

// ---------------- GAT softmax + aggregation: one wave per dst node ------------
__global__ void agg_kernel(const float* __restrict__ h, const float* __restrict__ als,
                           const float* __restrict__ ald, const int* __restrict__ start,
                           const int* __restrict__ indeg, const int* __restrict__ inedges,
                           const float* __restrict__ bias, _Float16* __restrict__ out16) {
    const int wave = threadIdx.x >> 5, lane = threadIdx.x & 31;
    const int node = blockIdx.x * 8 + wave;
    const int head = lane >> 3;                  // features lane*4..lane*4+3 all in one head
    const int beg = start[node], deg = indeg[node];
    const float ad = ald[node * HEADS + head];

    float m = -3.4e38f;
    for (int e = 0; e < deg; ++e) {
        const int s = inedges[beg + e];
        float l = als[s * HEADS + head] + ad;
        l = (l > 0.f) ? l : 0.2f * l;            // leaky_relu, NEG_SLOPE=0.2
        m = fmaxf(m, l);
    }
    float ssum = 0.f, a0 = 0.f, a1 = 0.f, a2 = 0.f, a3 = 0.f;
    for (int e = 0; e < deg; ++e) {
        const int s = inedges[beg + e];
        float l = als[s * HEADS + head] + ad;
        l = (l > 0.f) ? l : 0.2f * l;
        const float w = __expf(l - m);
        ssum += w;
        const float4 hv = *reinterpret_cast<const float4*>(h + (size_t)s * HID + lane * 4);
        a0 += w * hv.x; a1 += w * hv.y; a2 += w * hv.z; a3 += w * hv.w;
    }
    const float inv = 1.f / (ssum + 1e-16f);
    const int f = lane * 4;
    _Float16* op = out16 + (size_t)node * HID + f;
    op[0] = (_Float16)fmaxf(a0 * inv + bias[f + 0], 0.f);
    op[1] = (_Float16)fmaxf(a1 * inv + bias[f + 1], 0.f);
    op[2] = (_Float16)fmaxf(a2 * inv + bias[f + 2], 0.f);
    op[3] = (_Float16)fmaxf(a3 * inv + bias[f + 3], 0.f);
}

// ---------------- WMMA GEMM: Y = act(X(f16, Nn x KDIM) @ W + b) ----------------
// W is pre-packed (packw_kernel). Block: 128 threads = 4 waves, 64 rows/block.
// Per k-step, the packed 32xNP W chunk is copied into LDS with
// global_load_async_to_lds_b128 (ASYNCcnt), scattered to a 20-dword row pitch:
// rows are 16B-aligned (ds_load_b128) and the 16 lane addresses n*20 mod 64 are
// all distinct (conflict-free). A tiles load as 2x global_load_b128 per k-step.
// ACT: 0 = none, 1 = tanh.
template<int KDIM, int NOUT, int ACT, bool HAS_BIAS, bool OUT32, bool OUT16>
__global__ __launch_bounds__(128)
void gemm_kernel(const _Float16* __restrict__ X, const uint32_t* __restrict__ Wp,
                 const float* __restrict__ bias, float* __restrict__ Y32,
                 _Float16* __restrict__ Y16) {
    constexpr int NT = (NOUT + 15) / 16;     // N tiles per wave
    constexpr int NP = NT * 16;              // padded N
    constexpr int KS = KDIM / 32;            // K steps
    __shared__ __align__(16) uint32_t ldsb[NP * 20];

    const int tid = threadIdx.x;
    const int wave = tid >> 5, lane = tid & 31;
    const int hi = lane >> 4, lm = lane & 15;
    const int rowA = blockIdx.x * 64 + wave * 16 + lm;
    const uint32_t ldsBase = (uint32_t)(uintptr_t)(&ldsb[0]);

    v8f acc[NT] = {};

    for (int ks = 0; ks < KS; ++ks) {
        // Async-copy this k-step's packed chunk (NP*16 dwords) global -> LDS.
        // 16B chunk c covers (n = c/4, kp = 4*(c%4)..4*(c%4)+3).
        for (int c = tid; c < NP * 4; c += 128) {
            const int n = c >> 2, q = c & 3;
            const uint32_t* gp = Wp + (size_t)ks * NP * 16 + (size_t)c * 4;
            const uint32_t lofs = ldsBase + (uint32_t)(n * 80 + q * 16);
            asm volatile("global_load_async_to_lds_b128 %0, %1, off"
                         :: "v"(lofs), "v"(gp) : "memory");
        }
        asm volatile("s_wait_asynccnt 0x0" ::: "memory");
        __syncthreads();

        // A tile 16x32, ISA layout: lane lm is row M; VGPRs 0..3 hold K pairs
        // {0,2,4,6}+8*hi (16 contiguous bytes), VGPRs 4..7 hold {16,18,20,22}+8*hi.
        AV a;
        {
            const uint4* ap = reinterpret_cast<const uint4*>(
                X + (size_t)rowA * KDIM + ks * 32 + hi * 8);
            const uint4 A0 = ap[0];   // halves [base,    base+8)
            const uint4 A1 = ap[2];   // halves [base+16, base+24)
            a.u[0] = A0.x; a.u[1] = A0.y; a.u[2] = A0.z; a.u[3] = A0.w;
            a.u[4] = A1.x; a.u[5] = A1.y; a.u[6] = A1.z; a.u[7] = A1.w;
        }

#pragma unroll
        for (int t = 0; t < NT; ++t) {
            // B tile 32x16: lane lm is col N; K pairs {0,2,4,6}+8*hi then {16,18,20,22}+8*hi
            AV b;
#pragma unroll
            for (int v = 0; v < 8; ++v) {
                b.u[v] = ldsb[(t * 16 + lm) * 20 + v + hi * 4 + (v & 4)];
            }
            acc[t] = __builtin_amdgcn_wmma_f32_16x16x32_f16(
                false, a.v, false, b.v, (short)0, acc[t], false, false);
        }
        __syncthreads();
    }

    // Epilogue: D layout — lane lm is col; VGPR r is row r + 8*hi
#pragma unroll
    for (int t = 0; t < NT; ++t) {
        const int n = t * 16 + lm;
        if (n < NOUT) {
            float bv = 0.f;
            if constexpr (HAS_BIAS) bv = bias[n];
#pragma unroll
            for (int r = 0; r < 8; ++r) {
                const int grow = blockIdx.x * 64 + wave * 16 + r + hi * 8;
                float vv = acc[t][r];
                if constexpr (HAS_BIAS) vv += bv;
                if constexpr (ACT == 1) vv = tanhf(vv);
                if constexpr (OUT32) Y32[(size_t)grow * NOUT + n] = vv;
                if constexpr (OUT16) Y16[(size_t)grow * NOUT + n] = (_Float16)vv;
            }
        }
    }
}

// ---------------- driver ------------------------------------------------------
extern "C" void kernel_launch(void* const* d_in, const int* in_sizes, int n_in,
                              void* d_out, int out_size, void* d_ws, size_t ws_size,
                              hipStream_t stream) {
    (void)in_sizes; (void)n_in; (void)out_size; (void)ws_size;
    const float* obs = (const float*)d_in[0];
    const float* W1  = (const float*)d_in[1];
    const float* a1s = (const float*)d_in[2];
    const float* a1d = (const float*)d_in[3];
    const float* b1  = (const float*)d_in[4];
    const float* W2  = (const float*)d_in[5];
    const float* a2s = (const float*)d_in[6];
    const float* a2d = (const float*)d_in[7];
    const float* b2  = (const float*)d_in[8];
    const float* Wm1 = (const float*)d_in[9];
    const float* bm1 = (const float*)d_in[10];
    const float* Wm2 = (const float*)d_in[11];
    const float* bm2 = (const float*)d_in[12];
    float* out = (float*)d_out;

    char* ws = (char*)d_ws;
    size_t off = 0;
    auto alloc = [&](size_t bytes) -> char* {
        char* p = ws + off;
        off += (bytes + 255) & ~(size_t)255;
        return p;
    };
    int* nbr        = (int*)alloc((size_t)NTOT * KNB * 4);
    int* indeg      = (int*)alloc((size_t)NTOT * 4);
    int* start      = (int*)alloc((size_t)NTOT * 4);
    int* cursor     = (int*)alloc((size_t)NTOT * 4);
    int* inedges    = (int*)alloc((size_t)EDG * 4);
    int* bsum       = (int*)alloc(512 * 4);
    int* boff       = (int*)alloc(512 * 4);
    float* als      = (float*)alloc((size_t)NTOT * HEADS * 4);
    float* ald      = (float*)alloc((size_t)NTOT * HEADS * 4);
    _Float16* obs16 = (_Float16*)alloc((size_t)NTOT * DIN * 2);
    float* h        = (float*)alloc((size_t)NTOT * HID * 4);
    _Float16* act16 = (_Float16*)alloc((size_t)NTOT * HID * 2);
    _Float16* t16   = (_Float16*)alloc((size_t)NTOT * HID * 2);
    uint32_t* Wp1   = (uint32_t*)alloc((size_t)1 * 128 * 16 * 4);   // KS=1, NP=128
    uint32_t* Wp2   = (uint32_t*)alloc((size_t)4 * 128 * 16 * 4);   // KS=4, NP=128
    uint32_t* Wpm1  = (uint32_t*)alloc((size_t)4 * 128 * 16 * 4);   // KS=4, NP=128
    uint32_t* Wpm2  = (uint32_t*)alloc((size_t)4 * 16 * 16 * 4);    // KS=4, NP=16

    // Weight packing (tiny; independent of the graph build)
    packw_kernel<DIN, HID><<<8, 256, 0, stream>>>(W1, Wp1);
    packw_kernel<HID, HID><<<32, 256, 0, stream>>>(W2, Wp2);
    packw_kernel<HID, HID><<<32, 256, 0, stream>>>(Wm1, Wpm1);
    packw_kernel<HID, 8><<<4, 256, 0, stream>>>(Wm2, Wpm2);

    // Graph build (once; shared by both GAT layers)
    knn_kernel<<<BB, NNODE, 0, stream>>>(obs, nbr);
    hipMemsetAsync(indeg, 0, (size_t)NTOT * 4, stream);
    degcount_kernel<<<NTOT / 256, 256, 0, stream>>>(nbr, indeg);
    scan_block_kernel<<<NTOT / 256, 256, 0, stream>>>(indeg, start, bsum);
    scan_top_kernel<<<1, 512, 0, stream>>>(bsum, boff);
    scan_add_kernel<<<NTOT / 256, 256, 0, stream>>>(start, boff, cursor);
    fill_kernel<<<NTOT / 256, 256, 0, stream>>>(nbr, cursor, inedges);

    // GAT layer 1
    cvt_kernel<<<(NTOT * DIN) / 256, 256, 0, stream>>>(obs, obs16, NTOT * DIN);
    gemm_kernel<DIN, HID, 0, false, true, false><<<NTOT / 64, 128, 0, stream>>>(
        obs16, Wp1, nullptr, h, nullptr);
    al_kernel<<<(NTOT * HEADS) / 256, 256, 0, stream>>>(h, a1s, a1d, als, ald);
    agg_kernel<<<NTOT / 8, 256, 0, stream>>>(h, als, ald, start, indeg, inedges, b1, act16);

    // GAT layer 2
    gemm_kernel<HID, HID, 0, false, true, false><<<NTOT / 64, 128, 0, stream>>>(
        act16, Wp2, nullptr, h, nullptr);
    al_kernel<<<(NTOT * HEADS) / 256, 256, 0, stream>>>(h, a2s, a2d, als, ald);
    agg_kernel<<<NTOT / 8, 256, 0, stream>>>(h, als, ald, start, indeg, inedges, b2, act16);

    // MLP head: tanh(x @ Wm1 + bm1) @ Wm2 + bm2
    gemm_kernel<HID, HID, 1, true, false, true><<<NTOT / 64, 128, 0, stream>>>(
        act16, Wpm1, bm1, nullptr, t16);
    gemm_kernel<HID, 8, 0, true, true, false><<<NTOT / 64, 128, 0, stream>>>(
        t16, Wpm2, bm2, out, nullptr);
}
